// MultiHeadSelfAttention_89730456748648
// MI455X (gfx1250) — matrix operand order repaired and checked by reading, not compiled, not run on hardware
//
#include <hip/hip_runtime.h>
#include <hip/hip_bf16.h>
#include <stdint.h>

#define EMBED    1024
#define HEADS    16
#define HEAD_DIM 64
#define SEQ      2048
#define BATCH    2

typedef __attribute__((ext_vector_type(16))) _Float16 v16h;
typedef __attribute__((ext_vector_type(8)))  float    v8f;

union Tile {
    v16h     h;
    uint32_t u[8];
    uint4    q4[2];
};

__device__ __forceinline__ v8f vzero() {
    v8f z;
#pragma unroll
    for (int i = 0; i < 8; ++i) z[i] = 0.f;
    return z;
}

__device__ __forceinline__ v8f wmma16(const Tile& a, const Tile& b, v8f c) {
    // D = A(16x32 f16) * B(32x16 f16) + C(16x16 f32)
    return __builtin_amdgcn_wmma_f32_16x16x32_f16(
        false, a.h, false, b.h, (short)0, c, false, false);
}

// A operand tile: 16 rows (row = lane&15) x 32 k. p -> (row0, k0), ld in elements.
// Pair v holds k = 2v + 16g (contiguous)  ->  two b128 loads at +0 / +16 bytes.
__device__ __forceinline__ void load_tileA(const _Float16* __restrict__ p, int ld, Tile& t) {
    const int lane = threadIdx.x & 31;
    const _Float16* row = p + (size_t)(lane & 15) * ld + 16 * (lane >> 4);
#pragma unroll
    for (int v = 0; v < 4; ++v) t.u[v] = *(const uint32_t*)(row + 2 * v);
#pragma unroll
    for (int v = 4; v < 8; ++v) t.u[v] = *(const uint32_t*)(row + 2 * v);
}

// B operand tile: 16 cols (col = lane&15) x 32 k; lane-group g holds k = 16g..16g+15
// contiguous -> two b128 loads.
__device__ __forceinline__ void load_tileB(const _Float16* __restrict__ p, int ld, Tile& t) {
    const int lane = threadIdx.x & 31;
    const _Float16* row = p + (size_t)(lane & 15) * ld + 16 * (lane >> 4);
    t.q4[0] = *(const uint4*)(row);
    t.q4[1] = *(const uint4*)(row + 8);
}

__device__ __forceinline__ uint32_t pack2h(float a, float b) {
    union { _Float16 h[2]; uint32_t u; } x;
    x.h[0] = (_Float16)a;
    x.h[1] = (_Float16)b;
    return x.u;
}

__global__ void cvt_f32_to_f16(const float* __restrict__ src, _Float16* __restrict__ dst, int n) {
    int i = blockIdx.x * blockDim.x + threadIdx.x;
    int stride = gridDim.x * blockDim.x;
    for (; i < n; i += stride) dst[i] = (_Float16)src[i];
}

__global__ void build_maskbits(const int* __restrict__ mask, uint32_t* __restrict__ bits, int nwords) {
    int w = blockIdx.x * blockDim.x + threadIdx.x;
    if (w >= nwords) return;
    uint32_t b = 0;
    for (int j = 0; j < 32; ++j) b |= (mask[w * 32 + j] != 0 ? 1u : 0u) << j;
    bits[w] = b;
}

// Load one double-buffered operand set: 2 A tiles (m0, m0+16) + 4 B tiles (n0..n0+63)
__device__ __forceinline__ void load_set(const _Float16* __restrict__ Xp,
                                         const _Float16* __restrict__ Wp,
                                         int K, int k0, Tile (&a)[2], Tile (&b)[4]) {
    load_tileA(Xp + k0, K, a[0]);
    load_tileA(Xp + (size_t)16 * K + k0, K, a[1]);
#pragma unroll
    for (int j = 0; j < 4; ++j) load_tileB(Wp + (size_t)(16 * j) * K + k0, K, b[j]);
}

// Generic WMMA GEMM: out = (X[MxK] @ W[NtotxK]^T + bias) * outScale
// Wave computes a 32(M) x 64(N) tile, double-buffered over k (steps of 32, unrolled x2).
// Output address: addr = (m % Sdim)*sM + n*sN + (m / Sdim)*sB (supports transposed V store)
__global__ void __launch_bounds__(128)
gemm_wmma(const _Float16* __restrict__ X, const _Float16* __restrict__ W,
          const float* __restrict__ bias,
          _Float16* __restrict__ outH, float* __restrict__ outF,
          int M, int K, int Ntot,
          int Sdim, long sM, long sN, long sB, float outScale)
{
    const int wid  = threadIdx.x >> 5;
    const int lane = threadIdx.x & 31;
    const int m0 = blockIdx.y * 32;
    const int n0 = (blockIdx.x * 4 + wid) * 64;
    if (m0 >= M || n0 >= Ntot) return;

    v8f acc[2][4];
#pragma unroll
    for (int i = 0; i < 2; ++i)
#pragma unroll
        for (int j = 0; j < 4; ++j) acc[i][j] = vzero();

    const _Float16* Xp = X + (size_t)m0 * K;
    const _Float16* Wp = W + (size_t)n0 * K;

    Tile a0[2], b0[4], a1[2], b1[4];
    load_set(Xp, Wp, K, 0, a0, b0);          // prologue: set 0 @ k=0

    for (int k0 = 0; k0 < K; k0 += 64) {     // K is a multiple of 64
        load_set(Xp, Wp, K, k0 + 32, a1, b1);   // set 1 in flight while set 0 computes
#pragma unroll
        for (int i = 0; i < 2; ++i)
#pragma unroll
            for (int j = 0; j < 4; ++j) acc[i][j] = wmma16(a0[i], b0[j], acc[i][j]);
        if (k0 + 64 < K)
            load_set(Xp, Wp, K, k0 + 64, a0, b0); // set 0 in flight while set 1 computes
#pragma unroll
        for (int i = 0; i < 2; ++i)
#pragma unroll
            for (int j = 0; j < 4; ++j) acc[i][j] = wmma16(a1[i], b1[j], acc[i][j]);
    }

    const int nl = lane & 15, g = lane >> 4;
#pragma unroll
    for (int i = 0; i < 2; ++i)
#pragma unroll
        for (int j = 0; j < 4; ++j) {
            const int n = n0 + 16 * j + nl;
            const float bj = bias[n];
#pragma unroll
            for (int r = 0; r < 8; ++r) {
                const int m = m0 + 16 * i + r + 8 * g;
                const float val = (acc[i][j][r] + bj) * outScale;
                const long addr = (long)(m % Sdim) * sM + (long)n * sN + (long)(m / Sdim) * sB;
                if (outF) outF[addr] = val;
                else      outH[addr] = (_Float16)val;
            }
        }
}

// Flash attention, computed transposed:  S^T = K·Q^T,  O^T = V^T·P^T.
// One wave handles one (batch, head, 16-query tile); streams keys in blocks of 32.
__global__ void __launch_bounds__(128)
attention_wmma(const _Float16* __restrict__ Qh,  // [N,S,E] (pre-scaled by 1/8)
               const _Float16* __restrict__ Kh,  // [N,S,E]
               const _Float16* __restrict__ Vt,  // [N,E,S] (transposed)
               const uint32_t* __restrict__ maskbits, // [N, S/32]
               _Float16* __restrict__ Ctx)       // [N,S,E]
{
    const int wid  = threadIdx.x >> 5;
    const int lane = threadIdx.x & 31;
    const int w = blockIdx.x * 4 + wid;
    const int QT = SEQ / 16;
    const int qtile = w % QT;
    const int h  = (w / QT) % HEADS;
    const int nb = w / (QT * HEADS);

    const int q = lane & 15, g = lane >> 4;

    const _Float16* Qbase = Qh + ((size_t)nb * SEQ + (size_t)qtile * 16) * EMBED + h * HEAD_DIM;
    const _Float16* Kbase = Kh + (size_t)nb * SEQ * EMBED + h * HEAD_DIM;
    const _Float16* Vbase = Vt + (size_t)nb * EMBED * SEQ + (size_t)(h * HEAD_DIM) * SEQ;
    const uint32_t* mb    = maskbits + nb * (SEQ / 32);

    // Q^T B-operands for the two 32-wide d-chunks (resident for whole loop)
    Tile qb[2];
    load_tileB(Qbase + 0,  EMBED, qb[0]);
    load_tileB(Qbase + 32, EMBED, qb[1]);

    float mrun = -3.0e38f;
    float srun = 0.f;
    v8f o[4];
#pragma unroll
    for (int j = 0; j < 4; ++j) o[j] = vzero();

    for (int kb = 0; kb < SEQ; kb += 32) {
        const uint32_t mw = mb[kb >> 5];

        // Issue ALL loads for this block up front: K tiles first (needed first),
        // then V tiles (not needed until after softmax -> latency hidden).
        Tile ka[2][2], va[4];
#pragma unroll
        for (int t = 0; t < 2; ++t)
#pragma unroll
            for (int c = 0; c < 2; ++c)
                load_tileA(Kbase + (size_t)(kb + 16 * t) * EMBED + 32 * c, EMBED, ka[t][c]);
#pragma unroll
        for (int c2 = 0; c2 < 4; ++c2)
            load_tileA(Vbase + (size_t)(16 * c2) * SEQ + kb, SEQ, va[c2]);

        // Prefetch next key block's K/V cachelines (global_prefetch_b8).
        if (kb + 32 < SEQ) {
#pragma unroll
            for (int t = 0; t < 2; ++t)
                __builtin_prefetch(Kbase + (size_t)(kb + 32 + 16 * t + q) * EMBED, 0, 3);
#pragma unroll
            for (int c2 = 0; c2 < 4; ++c2)
                __builtin_prefetch(Vbase + (size_t)(16 * c2 + q) * SEQ + kb + 32, 0, 3);
        }

        // scores S^T: two 16(k) x 16(q) tiles, contraction over d = 64 (2 steps)
        v8f st[2];
        st[0] = vzero(); st[1] = vzero();
#pragma unroll
        for (int t = 0; t < 2; ++t)
#pragma unroll
            for (int c = 0; c < 2; ++c)
                st[t] = wmma16(ka[t][c], qb[c], st[t]);

        // mask + block row-max (per q, across lane pair q / q+16)
        float sc[2][8];
        float bmax = -3.0e38f;
#pragma unroll
        for (int t = 0; t < 2; ++t)
#pragma unroll
            for (int r = 0; r < 8; ++r) {
                const int bit = 16 * t + 8 * g + r;   // key offset within block
                float v = st[t][r];
                if (((mw >> bit) & 1u) == 0u) v = -3.0e38f;
                sc[t][r] = v;
                bmax = fmaxf(bmax, v);
            }
        bmax = fmaxf(bmax, __shfl_xor(bmax, 16, 32));
        const float mnew = fmaxf(mrun, bmax);
        const float corr = __expf(mrun - mnew);

        // exponentials, packed as f16 pairs; own lane covers k-offsets
        // {8g..8g+7} and {16+8g..16+8g+7}
        float psum = 0.f;
        uint32_t p0[4], p1[4];
#pragma unroll
        for (int jv = 0; jv < 4; ++jv) {
            const float a0 = __expf(sc[0][2 * jv]     - mnew);
            const float a1 = __expf(sc[0][2 * jv + 1] - mnew);
            const float b0 = __expf(sc[1][2 * jv]     - mnew);
            const float b1 = __expf(sc[1][2 * jv + 1] - mnew);
            psum += a0 + a1 + b0 + b1;
            p0[jv] = pack2h(a0, a1);
            p1[jv] = pack2h(b0, b1);
        }
        psum += __shfl_xor(psum, 16, 32);
        srun = srun * corr + psum;
#pragma unroll
        for (int j = 0; j < 4; ++j)
#pragma unroll
            for (int r = 0; r < 8; ++r) o[j][r] *= corr;

        // Build P^T B-operand: group g needs k = 16g..16g+15; fetch the missing
        // half from the partner lane (one shfl_xor(16) per packed pair).
        Tile pb;
#pragma unroll
        for (int jv = 0; jv < 4; ++jv) {
            const uint32_t o0 = (uint32_t)__shfl_xor((int)p0[jv], 16, 32);
            const uint32_t o1 = (uint32_t)__shfl_xor((int)p1[jv], 16, 32);
            pb.u[jv]     = (g == 0) ? p0[jv] : o1;
            pb.u[4 + jv] = (g == 0) ? o0 : p1[jv];
        }

        // O^T += V^T(16d x 32k) * P^T(32k x 16q), four d-chunks
#pragma unroll
        for (int c2 = 0; c2 < 4; ++c2)
            o[c2] = wmma16(va[c2], pb, o[c2]);

        mrun = mnew;
    }

    const float inv = 1.0f / srun;
    const size_t outRow = ((size_t)nb * SEQ + (size_t)qtile * 16 + q) * EMBED + h * HEAD_DIM;
#pragma unroll
    for (int c2 = 0; c2 < 4; ++c2)
#pragma unroll
        for (int r = 0; r < 8; ++r) {
            const int d = 16 * c2 + r + 8 * g;
            Ctx[outRow + d] = (_Float16)(o[c2][r] * inv);
        }
}

extern "C" void kernel_launch(void* const* d_in, const int* in_sizes, int n_in,
                              void* d_out, int out_size, void* d_ws, size_t ws_size,
                              hipStream_t stream) {
    (void)in_sizes; (void)n_in; (void)out_size; (void)ws_size;
    const float* x    = (const float*)d_in[0];
    const int*   mask = (const int*)d_in[1];
    const float* Wq   = (const float*)d_in[2];
    const float* bq   = (const float*)d_in[3];
    const float* Wk   = (const float*)d_in[4];
    const float* bk   = (const float*)d_in[5];
    const float* Wv   = (const float*)d_in[6];
    const float* bv   = (const float*)d_in[7];
    const float* Wo   = (const float*)d_in[8];
    const float* bo   = (const float*)d_in[9];
    float* out = (float*)d_out;

    const size_t nX = (size_t)BATCH * SEQ * EMBED;  // 4M
    const size_t nW = (size_t)EMBED * EMBED;        // 1M

    char* ws = (char*)d_ws;
    size_t off = 0;
    auto alloc = [&](size_t bytes) -> void* {
        void* p = ws + off;
        off += (bytes + 255) & ~(size_t)255;
        return p;
    };
    _Float16* xh   = (_Float16*)alloc(nX * 2);
    _Float16* wqh  = (_Float16*)alloc(nW * 2);
    _Float16* wkh  = (_Float16*)alloc(nW * 2);
    _Float16* wvh  = (_Float16*)alloc(nW * 2);
    _Float16* woh  = (_Float16*)alloc(nW * 2);
    _Float16* qbuf = (_Float16*)alloc(nX * 2);
    _Float16* kbuf = (_Float16*)alloc(nX * 2);
    _Float16* vtb  = (_Float16*)alloc(nX * 2);
    _Float16* ctx  = (_Float16*)alloc(nX * 2);
    uint32_t* mbits = (uint32_t*)alloc((size_t)BATCH * (SEQ / 32) * 4);

    cvt_f32_to_f16<<<1024, 256, 0, stream>>>(x,  xh,  (int)nX);
    cvt_f32_to_f16<<<512,  256, 0, stream>>>(Wq, wqh, (int)nW);
    cvt_f32_to_f16<<<512,  256, 0, stream>>>(Wk, wkh, (int)nW);
    cvt_f32_to_f16<<<512,  256, 0, stream>>>(Wv, wvh, (int)nW);
    cvt_f32_to_f16<<<512,  256, 0, stream>>>(Wo, woh, (int)nW);
    build_maskbits<<<1, 128, 0, stream>>>(mask, mbits, BATCH * SEQ / 32);

    const int M = BATCH * SEQ;                 // 4096
    dim3 gg(EMBED / 256, M / 32);              // (4, 128), 128-thread blocks (4 waves)
    const long SE = (long)SEQ * EMBED;

    // Q = (x@Wq^T + bq) * 1/sqrt(64), layout [N,S,E]
    gemm_wmma<<<gg, 128, 0, stream>>>(xh, wqh, bq, qbuf, nullptr,
                                      M, EMBED, EMBED, SEQ, (long)EMBED, 1L, SE, 0.125f);
    // K, layout [N,S,E]
    gemm_wmma<<<gg, 128, 0, stream>>>(xh, wkh, bk, kbuf, nullptr,
                                      M, EMBED, EMBED, SEQ, (long)EMBED, 1L, SE, 1.0f);
    // V, stored transposed: Vt[n][e][s]  (addr = s*1 + e*S + nb*E*S)
    gemm_wmma<<<gg, 128, 0, stream>>>(xh, wvh, bv, vtb, nullptr,
                                      M, EMBED, EMBED, SEQ, 1L, (long)SEQ, SE, 1.0f);
    // attention: one wave per (batch, head, 16-query tile)
    attention_wmma<<<BATCH * HEADS * (SEQ / 16) / 4, 128, 0, stream>>>(
        qbuf, kbuf, vtb, mbits, ctx);
    // out = ctx @ Wo^T + bo (fp32 output)
    gemm_wmma<<<gg, 128, 0, stream>>>(ctx, woh, bo, nullptr, out,
                                      M, EMBED, EMBED, SEQ, (long)EMBED, 1L, SE, 1.0f);
}
